// VectorQuantizer_21827023798476
// MI455X (gfx1250) — matrix operand order repaired and checked by reading, not compiled, not run on hardware
//
#include <hip/hip_runtime.h>
#include <hip/hip_bf16.h>
#include <math.h>

// ---------------------------------------------------------------------------
// VQ-VAE VectorQuantizer for MI455X (gfx1250, wave32, WMMA)
//   x:  [16,64,64,64] f32  -> flat [N=65536, D=64]
//   E:  [D=64, K=1024] f32
//   out = [ loss(1) | quantized(4194304) | perplexity(1) | encodings(67108864) ]
// Strategy: argmax_k ( x.e_k - 0.5||e_k||^2 ) via split-bf16 WMMA
//   (x_hi*e_hi + x_hi*e_lo + x_lo*e_hi), 3x v_wmma_f32_16x16x32_bf16 per
//   k-block.  B operand pre-packed into wave32 WMMA lane layout and staged
//   through LDS with double-buffered GLOBAL_LOAD_ASYNC_TO_LDS_B128.
// ---------------------------------------------------------------------------

typedef __attribute__((ext_vector_type(16))) __bf16 v16bf;
typedef __attribute__((ext_vector_type(8)))  float  v8f;
typedef __attribute__((ext_vector_type(4)))  float  f4;
typedef __attribute__((ext_vector_type(2)))  float  f2;

#define VQ_N      65536
#define VQ_D      64
#define VQ_K      1024
#define VQ_NELEM  4194304      // N*D
#define Q_OFF     1            // quantized offset in d_out (floats)
#define PERP_OFF  4194305
#define ENC_OFF   4194306

#define GRP  16                // staging groups
#define GCH  4                 // column chunks (of 16) per group

union U16x16 { unsigned short u[16]; v16bf v; };

__device__ __forceinline__ unsigned short bf16_rne(float f) {
    unsigned u = __float_as_uint(f);
    u += 0x7FFFu + ((u >> 16) & 1u);
    return (unsigned short)(u >> 16);
}
__device__ __forceinline__ void split_bf(float f, unsigned short& hi, unsigned short& lo) {
    hi = bf16_rne(f);
    float fh = __uint_as_float(((unsigned)hi) << 16);
    lo = bf16_rne(f - fh);
}

// CDNA5 async copy global -> LDS (ASYNCcnt); 16B per lane.
__device__ __forceinline__ void async_ld_b128(unsigned lds_addr, const void* gaddr) {
    asm volatile("global_load_async_to_lds_b128 %0, %1, off"
                 :: "v"(lds_addr), "v"(gaddr) : "memory");
}
__device__ __forceinline__ void wait_asynccnt0() {
    asm volatile("s_wait_asynccnt 0x0" ::: "memory");
}

// --- init: zero counts + loss accumulator ---------------------------------
__global__ void vq_init(int* counts, float* lossacc) {
    int t = threadIdx.x;
    if (t < VQ_K) counts[t] = 0;
    if (t == 0) *lossacc = 0.0f;
}

// --- pack codebook into WMMA B-operand layout, split hi/lo bf16 -----------
// Layout (ushort units): Bpk[cc*2048 + variant*1024 + kb*512 + lane*16 + slot]
//   cc: 16-column chunk (0..63), variant: 0=hi 1=lo, kb: K-block of 32 D's,
//   lane 0..15 -> column n=lane, D = kb*32 + 0..15 ; lane 16..31 -> D += 16
__global__ void vq_pack(const float* __restrict__ emb, unsigned short* __restrict__ Bpk) {
    int cc = blockIdx.x;
#pragma unroll
    for (int i = 0; i < 4; ++i) {
        int p    = threadIdx.x + i * 256;     // 0..1023
        int kb   = p >> 9;
        int lane = (p >> 4) & 31;
        int slot = p & 15;
        int hs   = lane >> 4;
        int n    = lane & 15;
        int d    = kb * 32 + hs * 16 + slot;
        float f  = emb[d * VQ_K + cc * 16 + n];
        unsigned short h, l;
        split_bf(f, h, l);
        Bpk[cc * 2048 +        kb * 512 + lane * 16 + slot] = h;
        Bpk[cc * 2048 + 1024 + kb * 512 + lane * 16 + slot] = l;
    }
}

// --- 0.5*||e_k||^2 per column ---------------------------------------------
__global__ void vq_colnorm(const float* __restrict__ emb, float* __restrict__ coln) {
    int k = blockIdx.x * 256 + threadIdx.x;
    float s = 0.0f;
#pragma unroll
    for (int d = 0; d < VQ_D; ++d) {
        float v = emb[d * VQ_K + k];
        s += v * v;
    }
    coln[k] = 0.5f * s;
}

// --- transpose codebook to [K, D] for coalesced gathers -------------------
__global__ void vq_embT(const float* __restrict__ emb, float* __restrict__ embT) {
    int gid = blockIdx.x * 256 + threadIdx.x;   // 65536
    int d = gid >> 10;
    int k = gid & 1023;
    embT[k * VQ_D + d] = emb[gid];              // gid == d*1024 + k (coalesced read)
}

// --- main: scores via split-bf16 WMMA, per-row argmax ---------------------
// 256 blocks x 512 threads (16 waves). Each wave owns 16 rows.
__global__ void __launch_bounds__(512)
vq_argmax(const float* __restrict__ x, const unsigned short* __restrict__ Bpk,
          const float* __restrict__ coln, int* __restrict__ idx_out) {
    __shared__ unsigned short sB[2][8192];      // 2 x 16 KB double buffer (4 chunks each)

    const int tid  = threadIdx.x;
    const int lane = tid & 31;
    const int wave = tid >> 5;
    const int hs   = lane >> 4;                 // half-select
    const int nloc = lane & 15;
    const int m0   = blockIdx.x * 256 + wave * 16;
    const int row  = m0 + nloc;                 // row for the A operand of this lane

    // ---- kick off async staging of group 0 while we build A operands
    {
        const char* gsrc = (const char*)Bpk + tid * 16;
        unsigned    ldst = (unsigned)(size_t)&sB[0][0] + tid * 16;
        async_ld_b128(ldst,        gsrc);
        async_ld_b128(ldst + 8192, gsrc + 8192);
    }

    // ---- build A operands (x rows) in WMMA 16x32 bf16 layout, hi/lo split
    U16x16 uh[2], ul[2];
    const float* xr = x + (size_t)row * VQ_D;
#pragma unroll
    for (int kb = 0; kb < 2; ++kb) {
#pragma unroll
        for (int t = 0; t < 8; ++t) {
            float f0 = xr[kb * 32 + hs * 16 + t];       // slots 0..7  -> K = kb*32 + hs*16 + t
            float f1 = xr[kb * 32 + hs * 16 + 8 + t];   // slots 8..15 -> K = kb*32 + hs*16 + 8 + t
            split_bf(f0, uh[kb].u[t],     ul[kb].u[t]);
            split_bf(f1, uh[kb].u[8 + t], ul[kb].u[8 + t]);
        }
    }
    v16bf Ah[2] = { uh[0].v, uh[1].v };
    v16bf Al[2] = { ul[0].v, ul[1].v };

    float best[8];
    int   bidx[8];
#pragma unroll
    for (int r = 0; r < 8; ++r) { best[r] = -3.4e38f; bidx[r] = 0; }

    for (int g = 0; g < GRP; ++g) {
        wait_asynccnt0();                       // our share of group g landed in LDS
        __syncthreads();                        // everyone's share landed / prev reads done
        const int buf = g & 1;

        if (g + 1 < GRP) {                      // prefetch next group into other buffer
            const char* gsrc = (const char*)Bpk + (g + 1) * 16384 + tid * 16;
            unsigned    ldst = (unsigned)(size_t)&sB[buf ^ 1][0] + tid * 16;
            async_ld_b128(ldst,        gsrc);
            async_ld_b128(ldst + 8192, gsrc + 8192);
        }

        const unsigned short* sBu = &sB[buf][0];
#pragma unroll
        for (int lc = 0; lc < GCH; ++lc) {
            int cc = g * GCH + lc;
            float cn = coln[cc * 16 + nloc];    // overlaps with WMMAs below

            v8f acc = {};                       // C = 0; norm folded in afterwards
#pragma unroll
            for (int kb = 0; kb < 2; ++kb) {
                v16bf bh = *(const v16bf*)(sBu + lc * 2048 +        kb * 512 + lane * 16);
                v16bf bl = *(const v16bf*)(sBu + lc * 2048 + 1024 + kb * 512 + lane * 16);
                acc = __builtin_amdgcn_wmma_f32_16x16x32_bf16(
                        false, Ah[kb], false, bh, (short)0, acc, false, false);
                acc = __builtin_amdgcn_wmma_f32_16x16x32_bf16(
                        false, Ah[kb], false, bl, (short)0, acc, false, false);
                acc = __builtin_amdgcn_wmma_f32_16x16x32_bf16(
                        false, Al[kb], false, bh, (short)0, acc, false, false);
            }

            int c = cc * 16 + nloc;             // column index held by this lane
#pragma unroll
            for (int r = 0; r < 8; ++r) {       // branch-free: keep EXEC full
                float s = acc[r] - cn;
                bool u = s > best[r];           // strict > => first index wins (ascending c)
                best[r] = u ? s : best[r];
                bidx[r] = u ? c : bidx[r];
            }
        }
    }

    // ---- butterfly argmax across the 16 lanes of each half (same M, varying N)
#pragma unroll
    for (int r = 0; r < 8; ++r) {
#pragma unroll
        for (int off = 1; off < 16; off <<= 1) {
            float ob = __shfl_xor(best[r], off, 32);
            int   oi = __shfl_xor(bidx[r], off, 32);
            bool u = (ob > best[r]) || (ob == best[r] && oi < bidx[r]);
            best[r] = u ? ob : best[r];
            bidx[r] = u ? oi : bidx[r];
        }
        if (nloc == 0)                          // lanes 0 and 16 write rows m0+r, m0+8+r
            idx_out[m0 + r + 8 * hs] = bidx[r];
    }
}

// --- quantized gather + MSE partial ---------------------------------------
__global__ void vq_quant(const float* __restrict__ x, const float* __restrict__ embT,
                         const int* __restrict__ idx, float* __restrict__ out,
                         float* __restrict__ lossacc) {
    __shared__ float red[256];
    int gid = blockIdx.x * 256 + threadIdx.x;   // 1,048,576 threads, 4 elems each
    int n  = gid >> 4;
    int dq = gid & 15;
    int id = idx[n];
    f4 q  = ((const f4*)(embT + (size_t)id * VQ_D))[dq];
    f4 xv = ((const f4*)x)[gid];
    float* o = out + Q_OFF + (size_t)gid * 4;   // 4B-aligned only: scalar stores
    __builtin_nontemporal_store(q.x, o);
    __builtin_nontemporal_store(q.y, o + 1);
    __builtin_nontemporal_store(q.z, o + 2);
    __builtin_nontemporal_store(q.w, o + 3);
    f4 dd = q - xv;
    red[threadIdx.x] = dd.x * dd.x + dd.y * dd.y + dd.z * dd.z + dd.w * dd.w;
    __syncthreads();
    for (int s = 128; s > 0; s >>= 1) {
        if (threadIdx.x < s) red[threadIdx.x] += red[threadIdx.x + s];
        __syncthreads();
    }
    if (threadIdx.x == 0) atomicAdd(lossacc, red[0]);
}

// --- code usage histogram --------------------------------------------------
__global__ void vq_hist(const int* __restrict__ idx, int* __restrict__ counts) {
    int n = blockIdx.x * 256 + threadIdx.x;
    atomicAdd(&counts[idx[n]], 1);
}

// --- one-hot encodings: 256 MB streaming store (nontemporal) ---------------
// One block per row: idx load is uniform -> scalar load.
__global__ void vq_onehot(const int* __restrict__ idx, float* __restrict__ enc) {
    int n  = blockIdx.x;
    int id = idx[n];                            // block-uniform: s_load
    int k0 = threadIdx.x * 4;
    f2 a, b;
    a.x = (k0     == id) ? 1.0f : 0.0f;
    a.y = (k0 + 1 == id) ? 1.0f : 0.0f;
    b.x = (k0 + 2 == id) ? 1.0f : 0.0f;
    b.y = (k0 + 3 == id) ? 1.0f : 0.0f;
    float* p = enc + (size_t)n * VQ_K + k0;     // base is 8B-aligned -> f2 stores
    __builtin_nontemporal_store(a, (f2*)p);
    __builtin_nontemporal_store(b, (f2*)(p + 2));
}

// --- finalize: loss + perplexity ------------------------------------------
__global__ void vq_final(const int* __restrict__ counts, const float* __restrict__ lossacc,
                         float* __restrict__ out) {
    __shared__ float red[256];
    float s = 0.0f;
    for (int k = threadIdx.x; k < VQ_K; k += 256) {
        float p = (float)counts[k] * (1.0f / (float)VQ_N);
        s += p * logf(p + 1e-10f);
    }
    red[threadIdx.x] = s;
    __syncthreads();
    for (int st = 128; st > 0; st >>= 1) {
        if (threadIdx.x < st) red[threadIdx.x] += red[threadIdx.x + st];
        __syncthreads();
    }
    if (threadIdx.x == 0) {
        out[0]        = 1.25f * (*lossacc) / (float)VQ_NELEM;  // q_loss + 0.25*e_loss
        out[PERP_OFF] = expf(-red[0]);
    }
}

// ---------------------------------------------------------------------------
extern "C" void kernel_launch(void* const* d_in, const int* in_sizes, int n_in,
                              void* d_out, int out_size, void* d_ws, size_t ws_size,
                              hipStream_t stream) {
    (void)in_sizes; (void)n_in; (void)out_size; (void)ws_size;
    const float* x   = (const float*)d_in[0];
    const float* emb = (const float*)d_in[1];
    float* out = (float*)d_out;

    char* ws = (char*)d_ws;
    unsigned short* Bpk   = (unsigned short*)(ws);            // 256 KB packed B hi/lo
    float*          coln  = (float*)(ws + 262144);            //   4 KB
    float*          embT  = (float*)(ws + 266240);            // 256 KB E^T
    int*            idx   = (int*)  (ws + 528384);            // 256 KB row indices
    int*            counts= (int*)  (ws + 790528);            //   4 KB
    float*          lossac= (float*)(ws + 794624);            //   4 B

    vq_init   <<<1,    1024, 0, stream>>>(counts, lossac);
    vq_pack   <<<64,    256, 0, stream>>>(emb, Bpk);
    vq_colnorm<<<4,     256, 0, stream>>>(emb, coln);
    vq_embT   <<<256,   256, 0, stream>>>(emb, embT);
    vq_argmax <<<256,   512, 0, stream>>>(x, Bpk, coln, idx);
    vq_quant  <<<4096,  256, 0, stream>>>(x, embT, idx, out, lossac);
    vq_hist   <<<256,   256, 0, stream>>>(idx, counts);
    vq_onehot <<<65536, 256, 0, stream>>>(idx, out + ENC_OFF);
    vq_final  <<<1,     256, 0, stream>>>(counts, lossac, out);
}
